// ODELSTMCell_58248346468499
// MI455X (gfx1250) — compile-verified, matrix operand order checked by reading, not run on hardware
//
#include <hip/hip_runtime.h>
#include <hip/hip_bf16.h>

typedef __bf16 bf16_t;
typedef bf16_t v16bf __attribute__((ext_vector_type(16)));
typedef bf16_t v8bf  __attribute__((ext_vector_type(8)));
typedef float  v8f   __attribute__((ext_vector_type(8)));

typedef unsigned int u32x4 __attribute__((ext_vector_type(4)));
typedef int i32x4 __attribute__((ext_vector_type(4)));
typedef int i32x8 __attribute__((ext_vector_type(8)));

#if defined(__has_builtin)
#if __has_builtin(__builtin_amdgcn_tensor_load_to_lds) && \
    __has_builtin(__builtin_amdgcn_s_wait_tensorcnt)
#define USE_TDM 1
#endif
#endif
#ifndef USE_TDM
#define USE_TDM 0
#endif

#if USE_TDM
// ---------------------------------------------------------------------------
// Tensor Data Mover: 2D tile load (bf16 elements) global -> LDS.
// D# per CDNA5 ISA:
//  group0: [1:0]=count=1, [63:32]=lds_addr, [120:64]=global_addr, [127:126]=2
//  group1: data_size=1 (2B), tensor_dim0/1, tile_dim0/1, tensor_dim0_stride
// Issued once per wave (EXEC ignored) -> call from one wave only; completion
// via s_wait_tensorcnt then workgroup barrier for visibility.
// ---------------------------------------------------------------------------
__device__ __forceinline__ void tdm_load_tile_2d(void* lds_dst,
                                                 const bf16_t* gsrc,
                                                 unsigned td0, unsigned td1,
                                                 unsigned tile0, unsigned tile1,
                                                 unsigned stride0) {
  const unsigned long long ga = (unsigned long long)(uintptr_t)gsrc;
  u32x4 g0;
  g0[0] = 1u;                                   // count=1 (valid user D#)
  g0[1] = (unsigned)(uintptr_t)lds_dst;         // lds_addr (LDS byte offset)
  g0[2] = (unsigned)ga;                         // global_addr[31:0]
  g0[3] = (unsigned)((ga >> 32) & 0x01FFFFFFull) | (2u << 30);  // type=2
  i32x8 g1;
  g1[0] = 0x00010000;                           // data_size = 2 bytes
  g1[1] = (int)((td0 & 0xFFFFu) << 16);         // tensor_dim0[15:0]
  g1[2] = (int)((td0 >> 16) | ((td1 & 0xFFFFu) << 16));
  g1[3] = (int)((td1 >> 16) | (tile0 << 16));   // tile_dim0
  g1[4] = (int)(tile1 & 0xFFFFu);               // tile_dim1 (tile_dim2=0)
  g1[5] = (int)stride0;                         // tensor_dim0_stride[31:0]
  g1[6] = 0;
  g1[7] = 0;
  const i32x4 z4 = {0, 0, 0, 0};
#if __clang_major__ >= 23
  const i32x8 z8 = {0, 0, 0, 0, 0, 0, 0, 0};
  __builtin_amdgcn_tensor_load_to_lds(g0, g1, z4, z4, z8, 0);
#else
  __builtin_amdgcn_tensor_load_to_lds(g0, g1, z4, z4, 0);
#endif
}
#endif  // USE_TDM

// ---------------------------------------------------------------------------
// WMMA fragment loader (CDNA5 ISA 7.12.2 layout, wave32): two ds_load_b128.
// ---------------------------------------------------------------------------
__device__ __forceinline__ v16bf frag_bf16(const bf16_t* __restrict__ base,
                                           int row_base, int ld, int k_base) {
  const int lane = threadIdx.x & 31;
  const int r = row_base + (lane & 15);
  const int kh = (lane >> 4) << 3;  // 0 or 8
  const bf16_t* p = base + r * ld + k_base + kh;
  const v8bf lo = *(const v8bf*)(p);
  const v8bf hi = *(const v8bf*)(p + 16);
  v16bf f;
#pragma unroll
  for (int i = 0; i < 8; ++i) f[i] = lo[i];
#pragma unroll
  for (int i = 0; i < 8; ++i) f[i + 8] = hi[i];
  return f;
}

__device__ __forceinline__ float sigmoidf_(float x) {
  return 1.0f / (1.0f + __expf(-x));
}

// ---------------------------------------------------------------------------
// Kernel 0: one-time weight conversion f32 -> bf16 (into workspace), so the
// 1024 GEMM blocks stage pure bf16 copies (or TDM DMAs) instead of each
// re-converting the same weights.
// ---------------------------------------------------------------------------
__global__ __launch_bounds__(256) void cvt_bf16_kernel(
    const float* __restrict__ src, bf16_t* __restrict__ dst, int n) {
  const int i = blockIdx.x * 256 + threadIdx.x;
  if (i < n) dst[i] = (bf16_t)src[i];
}

// ---------------------------------------------------------------------------
// Kernel 1: LSTMCell.  Block = 256 threads (8 waves), 16 rows, all 1024 gate
// columns.  gates = x@Wih^T + h@Whh^T (WMMA bf16, f32 accum), fused
// activations.  Weight K-chunks staged into LDS by the Tensor Data Mover.
// LDS bytes: a_x[0,8192) a_h[8192,16384) wih[16384,81920) whh[81920,147456)
//            gates f32 overlays wih at [16384,81920).  Total 147456.
// ---------------------------------------------------------------------------
__global__ __launch_bounds__(256) void lstm_kernel(
    const float* __restrict__ x, const float* __restrict__ hp,
    const float* __restrict__ cp,
    const bf16_t* __restrict__ WihB, const bf16_t* __restrict__ WhhB,
    const float* __restrict__ bih, const float* __restrict__ bhh,
    float* __restrict__ c_out, float* __restrict__ h_out) {
  extern __shared__ char smem[];
  bf16_t* a_x   = (bf16_t*)smem;                    // 16*256 bf16
  bf16_t* a_h   = (bf16_t*)(smem + 8192);           // 16*256 bf16
  bf16_t* wih   = (bf16_t*)(smem + 16384);          // 1024*32 bf16
  bf16_t* whh   = (bf16_t*)(smem + 81920);          // 1024*32 bf16
  float*  gates = (float*)(smem + 16384);           // 16*1024 f32 (overlay)

  const int tid = threadIdx.x;
  const int wv  = tid >> 5;
  const int m0  = blockIdx.x * 16;

  for (int i = tid; i < 16 * 256; i += 256) {
    const size_t g = (size_t)(m0 + (i >> 8)) * 256 + (i & 255);
    a_x[i] = (bf16_t)x[g];
    a_h[i] = (bf16_t)hp[g];
  }

  v8f acc[8];
#pragma unroll
  for (int j = 0; j < 8; ++j) acc[j] = {};

  for (int kc = 0; kc < 8; ++kc) {
    __syncthreads();  // previous chunk's WMMA reads done
    const int k0 = kc * 32;
#if USE_TDM
    if (wv == 0) {
      // 1024x32 bf16 K-chunk of each weight matrix, packed into LDS
      tdm_load_tile_2d(wih, WihB + k0, 256u, 1024u, 32u, 1024u, 256u);
      tdm_load_tile_2d(whh, WhhB + k0, 256u, 1024u, 32u, 1024u, 256u);
      __builtin_amdgcn_s_wait_tensorcnt(0);
    }
#else
    for (int i = tid; i < 1024 * 4; i += 256) {
      const int n = i >> 2, q = (i & 3) << 3;
      ((uint4*)wih)[i] = *(const uint4*)(WihB + (size_t)n * 256 + k0 + q);
      ((uint4*)whh)[i] = *(const uint4*)(WhhB + (size_t)n * 256 + k0 + q);
    }
    if (kc + 1 < 8) {
      const size_t pn = (size_t)(tid << 2) * 256 + k0 + 32;
      __builtin_prefetch(&WihB[pn], 0, 1);
      __builtin_prefetch(&WhhB[pn], 0, 1);
    }
#endif
    __syncthreads();
    const v16bf ax = frag_bf16(a_x, 0, 256, k0);
    const v16bf ah = frag_bf16(a_h, 0, 256, k0);
#pragma unroll
    for (int j = 0; j < 8; ++j) {
      const int nb = (wv * 8 + j) * 16;
      const v16bf bi = frag_bf16(wih, nb, 32, 0);
      const v16bf bh = frag_bf16(whh, nb, 32, 0);
      acc[j] = __builtin_amdgcn_wmma_f32_16x16x32_bf16(
          false, ax, false, bi, (short)0, acc[j], false, false);
      acc[j] = __builtin_amdgcn_wmma_f32_16x16x32_bf16(
          false, ah, false, bh, (short)0, acc[j], false, false);
    }
  }
  __syncthreads();
  {
    const int lane = tid & 31;
    const int col  = lane & 15;
    const int ro   = (lane >> 4) << 3;
#pragma unroll
    for (int j = 0; j < 8; ++j) {
      const int nb = (wv * 8 + j) * 16;
#pragma unroll
      for (int r = 0; r < 8; ++r)
        gates[(ro + r) * 1024 + nb + col] = acc[j][r];
    }
  }
  __syncthreads();
  for (int i = tid; i < 16 * 256; i += 256) {
    const int row = i >> 8, col = i & 255;
    const float gi = gates[row * 1024 + col]       + bih[col]       + bhh[col];
    const float gf = gates[row * 1024 + 256 + col] + bih[256 + col] + bhh[256 + col];
    const float gg = gates[row * 1024 + 512 + col] + bih[512 + col] + bhh[512 + col];
    const float go = gates[row * 1024 + 768 + col] + bih[768 + col] + bhh[768 + col];
    const float ig = sigmoidf_(gi);
    const float fg = sigmoidf_(gf);
    const float gt = tanhf(gg);
    const float og = sigmoidf_(go);
    const size_t g = (size_t)(m0 + row) * 256 + col;
    const float c  = fg * cp[g] + ig * gt;
    c_out[g] = c;
    h_out[g] = og * tanhf(c);
  }
}

// ---------------------------------------------------------------------------
// Kernel 2: fixed-step dopri5 neural ODE, persistent 16-row tile per block.
// Block = 128 threads (4 waves).  All state (y f32, stage input bf16, z bf16,
// k1..k6 f32) and both weight matrices stay LDS-resident across all 48 stage
// evaluations; weights arrive via one TDM DMA each.
// LDS bytes: w1l[0,32768) w2l[32768,65536) yb[65536,81920) yt[81920,90112)
//   zb[90112,92160) kb[92160,190464) b1l[190464) b2l[190720) dtl[191744).
//   Total 191808.
// ---------------------------------------------------------------------------
__constant__ float DP_A[6][5] = {
    {0.f, 0.f, 0.f, 0.f, 0.f},
    {0.2f, 0.f, 0.f, 0.f, 0.f},
    {3.f / 40.f, 9.f / 40.f, 0.f, 0.f, 0.f},
    {44.f / 45.f, -56.f / 15.f, 32.f / 9.f, 0.f, 0.f},
    {19372.f / 6561.f, -25360.f / 2187.f, 64448.f / 6561.f, -212.f / 729.f, 0.f},
    {9017.f / 3168.f, -355.f / 33.f, 46732.f / 5247.f, 49.f / 176.f,
     -5103.f / 18656.f},
};

__global__ __launch_bounds__(128) void ode_kernel(
    const float* __restrict__ y0, const float* __restrict__ ts,
    const bf16_t* __restrict__ W1B, const float* __restrict__ b1,
    const bf16_t* __restrict__ W2B, const float* __restrict__ b2,
    float* __restrict__ yout) {
  extern __shared__ char smem[];
  bf16_t* w1l = (bf16_t*)smem;                 // 64*256 bf16
  bf16_t* w2l = (bf16_t*)(smem + 32768);       // 256*64 bf16
  float*  yb  = (float*)(smem + 65536);        // 16*256 f32
  bf16_t* yt  = (bf16_t*)(smem + 81920);       // 16*256 bf16
  bf16_t* zb  = (bf16_t*)(smem + 90112);       // 16*64 bf16
  float*  kb  = (float*)(smem + 92160);        // 6*16*256 f32
  float*  b1l = (float*)(smem + 190464);       // 64
  float*  b2l = (float*)(smem + 190720);       // 256
  float*  dtl = (float*)(smem + 191744);       // 16

  const int tid  = threadIdx.x;
  const int wv   = tid >> 5;
  const int lane = tid & 31;
  const int m0   = blockIdx.x * 16;

#if USE_TDM
  if (wv == 0) {  // each matrix as one 16384-element bf16 row
    tdm_load_tile_2d(w1l, W1B, 16384u, 1u, 16384u, 1u, 16384u);
    tdm_load_tile_2d(w2l, W2B, 16384u, 1u, 16384u, 1u, 16384u);
    __builtin_amdgcn_s_wait_tensorcnt(0);
  }
#else
  for (int i = tid; i < 2048; i += 128) ((uint4*)w1l)[i] = ((const uint4*)W1B)[i];
  for (int i = tid; i < 2048; i += 128) ((uint4*)w2l)[i] = ((const uint4*)W2B)[i];
#endif
  if (tid < 64) b1l[tid] = b1[tid];
  for (int i = tid; i < 256; i += 128) b2l[i] = b2[i];
  if (tid < 16) dtl[tid] = ts[m0 + tid] * 0.125f;
  for (int i = tid; i < 16 * 256; i += 128) yb[i] = y0[(size_t)m0 * 256 + i];
  __syncthreads();

  for (int s = 0; s < 8; ++s) {
    for (int j = 0; j < 6; ++j) {
      // stage input: yt = bf16(yb + dt * sum_p A[j][p] * k_p)
      for (int i = tid * 2; i < 16 * 256; i += 256) {
        const float dt = dtl[i >> 8];
        float v0 = yb[i], v1 = yb[i + 1];
        for (int p = 0; p < 5; ++p) {
          const float a = DP_A[j][p];
          if (a != 0.0f) {
            const float c = dt * a;
            v0 += c * kb[p * 4096 + i];
            v1 += c * kb[p * 4096 + i + 1];
          }
        }
        yt[i]     = (bf16_t)v0;
        yt[i + 1] = (bf16_t)v1;
      }
      __syncthreads();
      // GEMM1: z = tanh(yt @ W1^T + b1)  (16x64, 1 tile per wave, K=256)
      {
        v8f acc = {};
#pragma unroll
        for (int kc = 0; kc < 8; ++kc) {
          const v16bf a = frag_bf16(yt, 0, 256, kc * 32);
          const v16bf b = frag_bf16(w1l, wv * 16, 256, kc * 32);
          acc = __builtin_amdgcn_wmma_f32_16x16x32_bf16(
              false, a, false, b, (short)0, acc, false, false);
        }
        const int col = wv * 16 + (lane & 15);
        const int ro  = (lane >> 4) << 3;
#pragma unroll
        for (int r = 0; r < 8; ++r)
          zb[(ro + r) * 64 + col] = (bf16_t)tanhf(acc[r] + b1l[col]);
      }
      __syncthreads();
      // GEMM2: k_j = z @ W2^T + b2  (16x256, 4 tiles per wave, K=64)
      float* kj = kb + j * 4096;
      const v16bf za0 = frag_bf16(zb, 0, 64, 0);   // shared across n-tiles
      const v16bf za1 = frag_bf16(zb, 0, 64, 32);
#pragma unroll
      for (int t = 0; t < 4; ++t) {
        const int nb = (wv + 4 * t) * 16;
        v8f acc = {};
        const v16bf b0 = frag_bf16(w2l, nb, 64, 0);
        acc = __builtin_amdgcn_wmma_f32_16x16x32_bf16(
            false, za0, false, b0, (short)0, acc, false, false);
        const v16bf b1f = frag_bf16(w2l, nb, 64, 32);
        acc = __builtin_amdgcn_wmma_f32_16x16x32_bf16(
            false, za1, false, b1f, (short)0, acc, false, false);
        const int col = nb + (lane & 15);
        const int ro  = (lane >> 4) << 3;
#pragma unroll
        for (int r = 0; r < 8; ++r)
          kj[(ro + r) * 256 + col] = acc[r] + b2l[col];
      }
      __syncthreads();
    }
    // dopri5 combine: y += dt*(35/384 k1 + 500/1113 k3 + 125/192 k4
    //                          - 2187/6784 k5 + 11/84 k6)
    for (int i = tid; i < 16 * 256; i += 128) {
      const float dt = dtl[i >> 8];
      yb[i] += dt * ((35.0f / 384.0f) * kb[i]
                   + (500.0f / 1113.0f) * kb[2 * 4096 + i]
                   + (125.0f / 192.0f) * kb[3 * 4096 + i]
                   + (-2187.0f / 6784.0f) * kb[4 * 4096 + i]
                   + (11.0f / 84.0f) * kb[5 * 4096 + i]);
    }
    __syncthreads();
  }
  for (int i = tid; i < 16 * 256; i += 128)
    yout[(size_t)m0 * 256 + i] = yb[i];
}

// ---------------------------------------------------------------------------
extern "C" void kernel_launch(void* const* d_in, const int* in_sizes, int n_in,
                              void* d_out, int out_size, void* d_ws,
                              size_t ws_size, hipStream_t stream) {
  const float* x   = (const float*)d_in[0];
  const float* hp  = (const float*)d_in[1];
  const float* cp  = (const float*)d_in[2];
  const float* ts  = (const float*)d_in[3];
  const float* Wih = (const float*)d_in[4];
  const float* Whh = (const float*)d_in[5];
  const float* bih = (const float*)d_in[6];
  const float* bhh = (const float*)d_in[7];
  const float* W1  = (const float*)d_in[8];
  const float* b1  = (const float*)d_in[9];
  const float* W2  = (const float*)d_in[10];
  const float* b2  = (const float*)d_in[11];

  const int Bn = 16384, Hn = 256;
  char* ws = (char*)d_ws;
  float*  h_ws = (float*)ws;                        // (B,H) f32      16 MB
  bf16_t* WihB = (bf16_t*)(ws + (size_t)16777216);  // 1024*256 bf16  512 KB
  bf16_t* WhhB = (bf16_t*)(ws + (size_t)17301504);  // 1024*256 bf16  512 KB
  bf16_t* W1B  = (bf16_t*)(ws + (size_t)17825792);  // 64*256 bf16     32 KB
  bf16_t* W2B  = (bf16_t*)(ws + (size_t)17858560);  // 256*64 bf16     32 KB

  float* yout = (float*)d_out;                    // h_ode (B,H)
  float* cout = (float*)d_out + (size_t)Bn * Hn;  // c     (B,H)

  cvt_bf16_kernel<<<1024, 256, 0, stream>>>(Wih, WihB, 1024 * 256);
  cvt_bf16_kernel<<<1024, 256, 0, stream>>>(Whh, WhhB, 1024 * 256);
  cvt_bf16_kernel<<<64, 256, 0, stream>>>(W1, W1B, 64 * 256);
  cvt_bf16_kernel<<<64, 256, 0, stream>>>(W2, W2B, 256 * 64);

  lstm_kernel<<<Bn / 16, 256, 147456, stream>>>(x, hp, cp, WihB, WhhB, bih, bhh,
                                                cout, h_ws);
  ode_kernel<<<Bn / 16, 128, 191808, stream>>>(h_ws, ts, W1B, b1, W2B, b2, yout);
}
